// VectorQuantizer_89395449298958
// MI455X (gfx1250) — matrix operand order repaired and checked by reading, not compiled, not run on hardware
//
#include <hip/hip_runtime.h>

// ---------------------------------------------------------------------------
// VQ-VAE vector quantizer for gfx1250 (MI455X).
// Distances via v_wmma_f32_16x16x32_bf16 (f32 accum); codebook staged to LDS
// with double-buffered async-to-LDS copies (ASYNCcnt); argmin fused.
// ---------------------------------------------------------------------------

typedef __attribute__((ext_vector_type(16))) __bf16 v16bf;
typedef __attribute__((ext_vector_type(8)))  float  v8f;
typedef __attribute__((ext_vector_type(4)))  int    v4i;
typedef __attribute__((address_space(3)))    v4i    lds_v4i;

#define KDIM   256
#define NCODES 16384
#define MROWS  8192     // 8 * 32 * 32
#define NCHUNK 64       // codes staged in LDS per iteration (32 KB)
#define NSPLIT 4        // N-dimension split across blockIdx.y
#define NELEM  2097152  // 8*256*32*32

union FragBF { v16bf v; uint4 u[2]; };

#if defined(__has_builtin)
#if __has_builtin(__builtin_amdgcn_global_load_async_to_lds_b128)
#define USE_ASYNC_LDS 1
#endif
#endif

__device__ inline void wait_async0() {
#if defined(__has_builtin) && __has_builtin(__builtin_amdgcn_s_wait_asynccnt)
  __builtin_amdgcn_s_wait_asynccnt(0);
#else
  asm volatile("s_wait_asynccnt 0x0" ::: "memory");
#endif
}

// ---------------------------------------------------------------------------
// Kernel 1: codebook f32 -> bf16 (row major) + per-code squared norm.
// ---------------------------------------------------------------------------
__global__ __launch_bounds__(256) void vq_prep_embed(
    const float* __restrict__ ew, __bf16* __restrict__ ebf,
    float* __restrict__ enorm) {
  const int wave = threadIdx.x >> 5;
  const int lane = threadIdx.x & 31;
  const int row  = blockIdx.x * 8 + wave;
  const float* src = ew + (size_t)row * KDIM + lane * 8;
  float4 f0 = ((const float4*)src)[0];
  float4 f1 = ((const float4*)src)[1];
  float ss = f0.x*f0.x + f0.y*f0.y + f0.z*f0.z + f0.w*f0.w
           + f1.x*f1.x + f1.y*f1.y + f1.z*f1.z + f1.w*f1.w;
  union { __bf16 h[8]; uint4 u; } pk;
  pk.h[0] = (__bf16)f0.x; pk.h[1] = (__bf16)f0.y;
  pk.h[2] = (__bf16)f0.z; pk.h[3] = (__bf16)f0.w;
  pk.h[4] = (__bf16)f1.x; pk.h[5] = (__bf16)f1.y;
  pk.h[6] = (__bf16)f1.z; pk.h[7] = (__bf16)f1.w;
  *(uint4*)(ebf + (size_t)row * KDIM + lane * 8) = pk.u;
  for (int off = 16; off > 0; off >>= 1) ss += __shfl_xor(ss, off, 32);
  if (lane == 0) enorm[row] = ss;
}

// ---------------------------------------------------------------------------
// Kernel 2: z (B,C,H,W) f32 -> zbf (B*H*W, C) bf16, LDS tile transpose.
// ---------------------------------------------------------------------------
__global__ __launch_bounds__(256) void vq_prep_z(
    const float* __restrict__ z, __bf16* __restrict__ zbf) {
  __shared__ float tile[32][33];
  const int b   = blockIdx.x >> 8;
  const int ct  = (blockIdx.x >> 5) & 7;
  const int ht  = blockIdx.x & 31;
  const int c0  = ct * 32, hw0 = ht * 32;
  const int tlo = threadIdx.x & 31, thi = threadIdx.x >> 5;
#pragma unroll
  for (int p = 0; p < 4; ++p) {
    int ci = p * 8 + thi;
    tile[ci][tlo] = z[(((size_t)b * 256 + c0 + ci) << 10) + hw0 + tlo];
  }
  __syncthreads();
#pragma unroll
  for (int p = 0; p < 4; ++p) {
    int hwi = p * 8 + thi;
    zbf[(size_t)((b << 10) + hw0 + hwi) * KDIM + c0 + tlo] =
        (__bf16)tile[tlo][hwi];
  }
}

// ---------------------------------------------------------------------------
// Stage one NCHUNK x KDIM bf16 tile (32 KB) from global into LDS.
// ---------------------------------------------------------------------------
__device__ inline void stage_chunk_async(const __bf16* __restrict__ src,
                                         __bf16* dst, int tid) {
#if defined(USE_ASYNC_LDS)
#pragma unroll
  for (int f = 0; f < (NCHUNK * KDIM / 8) / 256; ++f) {
    int e = (f * 256 + tid) * 8;
    __builtin_amdgcn_global_load_async_to_lds_b128(
        (v4i*)(src + e), (lds_v4i*)(dst + e), 0, 0);
  }
#else
  const uint4* s4 = (const uint4*)src;
  uint4* d4 = (uint4*)dst;
#pragma unroll
  for (int f = 0; f < (NCHUNK * KDIM / 8) / 256; ++f)
    d4[f * 256 + tid] = s4[f * 256 + tid];
#endif
}

// ---------------------------------------------------------------------------
// Kernel 3: distances + argmin via WMMA bf16.
// Block = 256 threads (8 waves). Wave owns 32 rows (two 16-row A tiles).
// gridDim = (32, NSPLIT): x -> M block of 256 rows, y -> N range of 4096.
// Double-buffered async LDS staging of 64-code tiles.
// ---------------------------------------------------------------------------
__global__ __launch_bounds__(256) void vq_argmin(
    const __bf16* __restrict__ zbf, const __bf16* __restrict__ ebf,
    const float* __restrict__ enorm, float* __restrict__ bestS_out,
    int* __restrict__ bestI_out) {
  __shared__ alignas(16) __bf16 tile[2][NCHUNK * KDIM];  // 2 x 32 KB

  const int tid  = threadIdx.x;
  const int wave = tid >> 5;
  const int lane = tid & 31;
  const int l16  = lane & 15;
  const int hi   = lane >> 4;
  const int m0   = blockIdx.x * 256 + wave * 32;
  const int ny   = blockIdx.y;
  const int nLo  = ny * (NCODES / NSPLIT);
  const int nChunks = (NCODES / NSPLIT) / NCHUNK;  // 64

  // A fragments for two M tiles; ISA 16-bit A 16x32 layout:
  // lane<16: elems 0..7 <- K base+0..7, elems 8..15 <- K base+16..23;
  // lanes>=16 shifted by +8.
  FragBF a[2][8];
#pragma unroll
  for (int mt = 0; mt < 2; ++mt) {
    const __bf16* zrow = zbf + (size_t)(m0 + mt * 16 + l16) * KDIM;
#pragma unroll
    for (int t = 0; t < 8; ++t) {
      int k0 = 32 * t + 8 * hi;
      a[mt][t].u[0] = *(const uint4*)(zrow + k0);
      a[mt][t].u[1] = *(const uint4*)(zrow + k0 + 16);
    }
  }

  float best[2][8];
  int   bidx[2][8];
#pragma unroll
  for (int mt = 0; mt < 2; ++mt)
#pragma unroll
    for (int v = 0; v < 8; ++v) { best[mt][v] = 3.4e38f; bidx[mt][v] = 0; }

  // Prologue: stage chunk 0 into buffer 0.
  stage_chunk_async(ebf + (size_t)nLo * KDIM, tile[0], tid);
  wait_async0();
  __syncthreads();

  for (int ch = 0; ch < nChunks; ++ch) {
    const __bf16* cur = tile[ch & 1];
    // Kick off the next chunk into the other buffer (overlaps compute).
    if (ch + 1 < nChunks)
      stage_chunk_async(ebf + (size_t)(nLo + (ch + 1) * NCHUNK) * KDIM,
                        tile[(ch + 1) & 1], tid);

    const int nBase = nLo + ch * NCHUNK;
    // 4 N-subtiles of 16 codes as two interleaved pairs; each B fragment
    // feeds both M tiles -> 4 independent WMMA chains, half the LDS reads.
#pragma unroll
    for (int sp = 0; sp < NCHUNK / 32; ++sp) {
      v8f acc00 = {}, acc01 = {}, acc10 = {}, acc11 = {};
      // B layout: N = lane%16, elems 0..7 <- K 0..7, 8..15 <- K 16..23 (+8 for lanes>=16).
      const __bf16* b0 = cur + (size_t)(sp * 32 + l16) * KDIM + 8 * hi;
      const __bf16* b1 = b0 + 16 * KDIM;
#pragma unroll
      for (int t = 0; t < 8; ++t) {
        FragBF fb0, fb1;
        fb0.u[0] = *(const uint4*)(b0 + 32 * t);
        fb0.u[1] = *(const uint4*)(b0 + 32 * t + 16);
        fb1.u[0] = *(const uint4*)(b1 + 32 * t);
        fb1.u[1] = *(const uint4*)(b1 + 32 * t + 16);
        acc00 = __builtin_amdgcn_wmma_f32_16x16x32_bf16(
            false, a[0][t].v, false, fb0.v, (short)0, acc00, false, false);
        acc10 = __builtin_amdgcn_wmma_f32_16x16x32_bf16(
            false, a[1][t].v, false, fb0.v, (short)0, acc10, false, false);
        acc01 = __builtin_amdgcn_wmma_f32_16x16x32_bf16(
            false, a[0][t].v, false, fb1.v, (short)0, acc01, false, false);
        acc11 = __builtin_amdgcn_wmma_f32_16x16x32_bf16(
            false, a[1][t].v, false, fb1.v, (short)0, acc11, false, false);
      }
      const int n0 = nBase + sp * 32 + l16;
      const int n1 = n0 + 16;
      const float e0 = enorm[n0];
      const float e1 = enorm[n1];
#pragma unroll
      for (int v = 0; v < 8; ++v) {
        float s00 = e0 - 2.0f * acc00[v];
        float s01 = e1 - 2.0f * acc01[v];
        float s10 = e0 - 2.0f * acc10[v];
        float s11 = e1 - 2.0f * acc11[v];
        if (s00 < best[0][v]) { best[0][v] = s00; bidx[0][v] = n0; }
        if (s01 < best[0][v]) { best[0][v] = s01; bidx[0][v] = n1; }
        if (s10 < best[1][v]) { best[1][v] = s10; bidx[1][v] = n0; }
        if (s11 < best[1][v]) { best[1][v] = s11; bidx[1][v] = n1; }
      }
    }

    // Close the pipeline stage: our async copies done, all waves done with cur.
    wait_async0();
    __syncthreads();
  }

  // Cross-lane argmin within the 16 lanes sharing the same rows.
  // C/D layout: row M = v + 8*hi, col N = lane%16.
#pragma unroll
  for (int mt = 0; mt < 2; ++mt) {
#pragma unroll
    for (int v = 0; v < 8; ++v) {
      float bs = best[mt][v];
      int   bi = bidx[mt][v];
#pragma unroll
      for (int off = 1; off < 16; off <<= 1) {
        float os = __shfl_xor(bs, off, 32);
        int   oi = __shfl_xor(bi, off, 32);
        if (os < bs || (os == bs && oi < bi)) { bs = os; bi = oi; }
      }
      if (l16 == 0) {
        int row = m0 + mt * 16 + 8 * hi + v;
        bestS_out[(size_t)ny * MROWS + row] = bs;
        bestI_out[(size_t)ny * MROWS + row] = bi;
      }
    }
  }
}

// ---------------------------------------------------------------------------
// Kernel 4: merge the NSPLIT candidates per row (first-index tie-break).
// ---------------------------------------------------------------------------
__global__ __launch_bounds__(256) void vq_merge(
    const float* __restrict__ bestS, const int* __restrict__ bestI,
    int* __restrict__ idx) {
  const int r = blockIdx.x * 256 + threadIdx.x;
  float bs = bestS[r];
  int   bi = bestI[r];
#pragma unroll
  for (int s = 1; s < NSPLIT; ++s) {
    float os = bestS[(size_t)s * MROWS + r];
    int   oi = bestI[(size_t)s * MROWS + r];
    if (os < bs || (os == bs && oi < bi)) { bs = os; bi = oi; }
  }
  idx[r] = bi;
}

// ---------------------------------------------------------------------------
// Kernel 5: z_q = embed_w[idx] in (B,C,H,W) order + block partial loss sums.
// ---------------------------------------------------------------------------
__global__ __launch_bounds__(256) void vq_gather_loss(
    const float* __restrict__ z, const float* __restrict__ ew,
    const int* __restrict__ idx, float* __restrict__ zq,
    float* __restrict__ partial) {
  __shared__ float sm[256];
  const size_t o = (size_t)blockIdx.x * 256 + threadIdx.x;
  const int b  = (int)(o >> 18);
  const int c  = (int)((o >> 10) & 255);
  const int hw = (int)(o & 1023);
  const int n  = idx[(b << 10) + hw];
  const float q = ew[(size_t)n * KDIM + c];
  zq[o] = q;
  const float d = q - z[o];
  sm[threadIdx.x] = d * d;
  __syncthreads();
  for (int s = 128; s > 0; s >>= 1) {
    if (threadIdx.x < s) sm[threadIdx.x] += sm[threadIdx.x + s];
    __syncthreads();
  }
  if (threadIdx.x == 0) partial[blockIdx.x] = sm[0];
}

// ---------------------------------------------------------------------------
// Kernel 6: deterministic final reduction; loss = (1+beta)*mean, beta = 1.
// ---------------------------------------------------------------------------
__global__ __launch_bounds__(256) void vq_finalize(
    const float* __restrict__ partial, float* __restrict__ loss) {
  __shared__ float sm[256];
  float s = 0.0f;
  for (int i = threadIdx.x; i < MROWS; i += 256) s += partial[i];
  sm[threadIdx.x] = s;
  __syncthreads();
  for (int t = 128; t > 0; t >>= 1) {
    if (threadIdx.x < t) sm[threadIdx.x] += sm[threadIdx.x + t];
    __syncthreads();
  }
  if (threadIdx.x == 0) *loss = 2.0f * sm[0] / (float)NELEM;
}

// ---------------------------------------------------------------------------
extern "C" void kernel_launch(void* const* d_in, const int* in_sizes, int n_in,
                              void* d_out, int out_size, void* d_ws,
                              size_t ws_size, hipStream_t stream) {
  (void)in_sizes; (void)n_in; (void)out_size; (void)ws_size;
  const float* z  = (const float*)d_in[0];   // (8,256,32,32)
  const float* ew = (const float*)d_in[1];   // (16384,256)
  float* out = (float*)d_out;                // z_q flat (2097152) + loss (1)

  // Workspace layout (bytes).
  char* ws = (char*)d_ws;
  size_t off = 0;
  __bf16* ebf  = (__bf16*)(ws + off); off += (size_t)NCODES * KDIM * 2;  // 8 MB
  __bf16* zbf  = (__bf16*)(ws + off); off += (size_t)MROWS * KDIM * 2;   // 4 MB
  float* enorm = (float*)(ws + off);  off += (size_t)NCODES * 4;         // 64 KB
  float* bestS = (float*)(ws + off);  off += (size_t)NSPLIT * MROWS * 4; // 128 KB
  int*   bestI = (int*)(ws + off);    off += (size_t)NSPLIT * MROWS * 4; // 128 KB
  int*   idx   = (int*)(ws + off);    off += (size_t)MROWS * 4;          // 32 KB
  float* part  = (float*)(ws + off);  off += (size_t)NELEM / 256 * 4;    // 32 KB

  vq_prep_embed<<<NCODES / 8, 256, 0, stream>>>(ew, ebf, enorm);
  vq_prep_z<<<2048, 256, 0, stream>>>(z, zbf);
  vq_argmin<<<dim3(MROWS / 256, NSPLIT), 256, 0, stream>>>(zbf, ebf, enorm,
                                                           bestS, bestI);
  vq_merge<<<MROWS / 256, 256, 0, stream>>>(bestS, bestI, idx);
  vq_gather_loss<<<NELEM / 256, 256, 0, stream>>>(z, ew, idx, out, part);
  vq_finalize<<<1, 256, 0, stream>>>(part, out + NELEM);
}
